// MultiHeadAttention_lora_34969623724625
// MI455X (gfx1250) — compile-verified, matrix operand order checked by reading, not run on hardware
//
#include <hip/hip_runtime.h>

// ---------- problem constants (from reference) ----------
#define B_  2
#define S_  2048
#define D_  768
#define H_  12
#define DK_ 64
#define R_  8

typedef __attribute__((ext_vector_type(16))) __bf16        v16bf;
typedef __attribute__((ext_vector_type(8)))  float         v8f;
typedef __attribute__((ext_vector_type(4)))  unsigned int  u32x4;
typedef __attribute__((ext_vector_type(8)))  int           i32x8;
typedef __attribute__((ext_vector_type(4)))  int           i32x4;

union FragU { v16bf v; u32x4 q[2]; };

// round-to-nearest-even f32 -> bf16 (stored as raw ushort)
__device__ __forceinline__ unsigned short f2bf(float f) {
  unsigned u = __float_as_uint(f);
  u += 0x7fffu + ((u >> 16) & 1u);
  return (unsigned short)(u >> 16);
}

// Load a 16-bit WMMA A/B fragment (16x32 or 32x16) for this lane.
// Per CDNA5 ISA layout: lanes 0-15 hold K = kbase+[0..7] in VGPR0-3 and
// K = kbase+16+[0..7] in VGPR4-7; lanes 16-31 the +8 halves.
// `row` points at the start of this lane's matrix row (contiguous in K).
__device__ __forceinline__ v16bf load_frag(const unsigned short* row, int kbase, int half) {
  FragU u;
  u.q[0] = *(const u32x4*)(row + kbase + 8 * half);
  u.q[1] = *(const u32x4*)(row + kbase + 16 + 8 * half);
  return u.v;
}

__device__ __forceinline__ v8f wmma_bf16(v16bf a, v16bf b, v8f c) {
  return __builtin_amdgcn_wmma_f32_16x16x32_bf16(false, a, false, b, (short)0, c, false, false);
}

// ---------------------------------------------------------------------------
// TDM: issue a 2D bf16 tile load (tile_w x tile_h elems, row stride in elems)
// from global memory into LDS.  Descriptor per CDNA5 ISA §8 (D# groups 0/1).
// Tensor dims are set equal to tile dims so every access is in-bounds.
// ---------------------------------------------------------------------------
__device__ __forceinline__ void tdm_load_2d(unsigned lds_off, unsigned long long gaddr,
                                            unsigned tile_w, unsigned tile_h,
                                            unsigned stride_elems) {
  u32x4 g0;
  g0[0] = 1u;                                              // count=1, user mode
  g0[1] = lds_off;                                         // lds_addr (bytes)
  g0[2] = (unsigned)(gaddr & 0xffffffffull);               // global_addr[95:64]
  g0[3] = (unsigned)((gaddr >> 32) & 0x01ffffffull)        // global_addr[120:96]
          | 0x80000000u;                                   // type=2 ("image")
  i32x8 g1;
  g1[0] = 0x10000;                                         // data_size=1 (2 bytes)
  g1[1] = (int)((tile_w & 0xffffu) << 16);                 // tensor_dim0 lo16
  g1[2] = (int)((tile_w >> 16) | ((tile_h & 0xffffu) << 16)); // dim0 hi | tensor_dim1 lo
  g1[3] = (int)((tile_h >> 16) | ((tile_w & 0xffffu) << 16)); // dim1 hi | tile_dim0
  g1[4] = (int)(tile_h & 0xffffu);                         // tile_dim1 ; tile_dim2=0
  g1[5] = (int)stride_elems;                               // tensor_dim0_stride lo32
  g1[6] = 0;
  g1[7] = 0;
  i32x4 z4;
  z4[0] = z4[1] = z4[2] = z4[3] = 0;
  i32x8 z8;
#pragma unroll
  for (int e = 0; e < 8; ++e) z8[e] = 0;
  __builtin_amdgcn_tensor_load_to_lds(g0, g1, z4, z4, z8, 0);
}

// ---------------------------------------------------------------------------
// f32 -> bf16 bulk convert
// ---------------------------------------------------------------------------
__global__ __launch_bounds__(256) void convert_f32_bf16(const float* __restrict__ in,
                                                        unsigned short* __restrict__ out,
                                                        size_t n) {
  size_t stride = (size_t)gridDim.x * 256;
  for (size_t i = (size_t)blockIdx.x * 256 + threadIdx.x; i < n; i += stride)
    out[i] = f2bf(in[i]);
}

// ---------------------------------------------------------------------------
// LoRA adapt: out = bf16( x + (x @ A) @ Bm ), one block per row of x [*, D]
// ---------------------------------------------------------------------------
__global__ __launch_bounds__(256) void lora_adapt(const float* __restrict__ x,
                                                  const float* __restrict__ A,
                                                  const float* __restrict__ Bm,
                                                  unsigned short* __restrict__ out) {
  __shared__ float red[256][R_];
  const int tid = threadIdx.x;
  const float* xr = x + (size_t)blockIdx.x * D_;

  float part[R_];
#pragma unroll
  for (int r = 0; r < R_; ++r) part[r] = 0.f;
  for (int d = tid; d < D_; d += 256) {
    float xv = xr[d];
#pragma unroll
    for (int r = 0; r < R_; ++r) part[r] += xv * A[d * R_ + r];
  }
#pragma unroll
  for (int r = 0; r < R_; ++r) red[tid][r] = part[r];
  __syncthreads();
  for (int s = 128; s > 0; s >>= 1) {
    if (tid < s) {
#pragma unroll
      for (int r = 0; r < R_; ++r) red[tid][r] += red[tid + s][r];
    }
    __syncthreads();
  }
  float t[R_];
#pragma unroll
  for (int r = 0; r < R_; ++r) t[r] = red[0][r];

  unsigned short* orow = out + (size_t)blockIdx.x * D_;
  for (int j = tid; j < D_; j += 256) {
    float v = xr[j];
#pragma unroll
    for (int r = 0; r < R_; ++r) v += t[r] * Bm[r * D_ + j];
    orow[j] = f2bf(v);
  }
}

// ---------------------------------------------------------------------------
// bf16 WMMA GEMM:  out = X[M,K] @ Wt[N,K]^T + bias[N]
//   wave tile 32(M) x 64(N), block = 8 waves = 64(M) x 256(N)
// MODE 0: plain f32 out [M,N]
// MODE 1: bf16 head-split out  [B,H,S,DK]  (n = h*64+d, m = b*S+s)
// MODE 2: bf16 head-split-T    [B,H,DK,S]
// ---------------------------------------------------------------------------
template <int MODE>
__global__ __launch_bounds__(256) void gemm_bf16_wmma(const unsigned short* __restrict__ X,
                                                      const unsigned short* __restrict__ Wt,
                                                      const float* __restrict__ bias,
                                                      void* __restrict__ outp,
                                                      int M, int N, int K) {
  const int lane = threadIdx.x & 31;
  const int w    = threadIdx.x >> 5;
  const int wm   = w & 1;
  const int wn   = w >> 1;
  const int half = lane >> 4;
  const int nn   = lane & 15;
  const int m0 = blockIdx.x * 64 + wm * 32;
  const int n0 = blockIdx.y * 256 + wn * 64;

  v8f zero;
#pragma unroll
  for (int e = 0; e < 8; ++e) zero[e] = 0.f;
  v8f acc[2][4];
#pragma unroll
  for (int i = 0; i < 2; ++i)
#pragma unroll
    for (int j = 0; j < 4; ++j) acc[i][j] = zero;

  const unsigned short* xrow0 = X + (size_t)(m0 + nn) * K;
  const unsigned short* xrow1 = X + (size_t)(m0 + 16 + nn) * K;
  const unsigned short* wrow[4];
#pragma unroll
  for (int j = 0; j < 4; ++j) wrow[j] = Wt + (size_t)(n0 + 16 * j + nn) * K;

#pragma unroll 3
  for (int k0 = 0; k0 < K; k0 += 32) {
    v16bf a0 = load_frag(xrow0, k0, half);
    v16bf a1 = load_frag(xrow1, k0, half);
#pragma unroll
    for (int j = 0; j < 4; ++j) {
      v16bf bb = load_frag(wrow[j], k0, half);
      acc[0][j] = wmma_bf16(a0, bb, acc[0][j]);
      acc[1][j] = wmma_bf16(a1, bb, acc[1][j]);
    }
  }

#pragma unroll
  for (int i = 0; i < 2; ++i) {
#pragma unroll
    for (int j = 0; j < 4; ++j) {
      const int n = n0 + 16 * j + nn;
      const float bn = bias[n];
#pragma unroll
      for (int r = 0; r < 8; ++r) {
        const int m = m0 + 16 * i + r + 8 * half;
        const float val = acc[i][j][r] + bn;
        if (MODE == 0) {
          ((float*)outp)[(size_t)m * N + n] = val;
        } else {
          const int b = m / S_, s = m % S_;
          const int hh = n >> 6, dd = n & 63;
          unsigned short* o = (unsigned short*)outp;
          if (MODE == 1)
            o[(((size_t)(b * H_ + hh) * S_) + s) * DK_ + dd] = f2bf(val);
          else
            o[(((size_t)(b * H_ + hh) * DK_) + dd) * S_ + s] = f2bf(val);
        }
      }
    }
  }
}

// ---------------------------------------------------------------------------
// Fused flash attention (bf16 WMMA, f32 accumulate, online softmax).
//   qh,kh: [B*H, S, DK] bf16 ; vT: [B*H, DK, S] bf16 ; mask: [B,S,S] i32
//   ctx out: [B, S, D] bf16 (head-merged)
// 1 wave = 16 query rows; block = 8 waves = 128 rows.
// K/V tiles are DMA'd into LDS by the Tensor Data Mover (wave 0, double
// buffered, TENSORcnt-tracked); all 8 waves consume them via ds_load_b128.
// ---------------------------------------------------------------------------
__global__ __launch_bounds__(256) void attn_wmma(const unsigned short* __restrict__ qh,
                                                 const unsigned short* __restrict__ kh,
                                                 const unsigned short* __restrict__ vT,
                                                 const int* __restrict__ mask,
                                                 unsigned short* __restrict__ ctx) {
  __shared__ __align__(16) unsigned short kbuf[2][32 * DK_];  // K tile: 32 rows x 64
  __shared__ __align__(16) unsigned short vbuf[2][DK_ * 32];  // V tile: 64 rows x 32
  __shared__ __align__(16) unsigned short plds[8][16 * 32];   // per-wave P tile

  const int lane = threadIdx.x & 31;
  const int w    = threadIdx.x >> 5;
  const int half = lane >> 4;
  const int nn   = lane & 15;
  const int bh = blockIdx.y;
  const int b = bh / H_, h = bh % H_;
  const int qr0 = blockIdx.x * 128 + w * 16;

  const unsigned short* qbase = qh + ((size_t)bh * S_ + qr0) * DK_;
  const unsigned short* kbase = kh + (size_t)bh * S_ * DK_;
  const unsigned short* vbase = vT + (size_t)bh * DK_ * S_;

  // Q A-fragments for this wave's 16 rows (K = 0..31 and 32..63), kept live.
  const unsigned short* qrow = qbase + (size_t)nn * DK_;
  v16bf qa0 = load_frag(qrow, 0, half);
  v16bf qa1 = load_frag(qrow, 32, half);

  v8f zero;
#pragma unroll
  for (int e = 0; e < 8; ++e) zero[e] = 0.f;
  v8f o[4];
#pragma unroll
  for (int c = 0; c < 4; ++c) o[c] = zero;

  float mrow[8], lrow[8];
  const int* mrowp[8];
#pragma unroll
  for (int r = 0; r < 8; ++r) {
    mrow[r] = -3.0e38f;
    lrow[r] = 0.f;
    mrowp[r] = mask + (size_t)b * S_ * S_ + (size_t)(qr0 + r + 8 * half) * S_;
  }

  // ---- TDM prologue: stage first K/V tiles into buffer 0 ----
  if (w == 0) {
    tdm_load_2d((unsigned)(size_t)(void*)&kbuf[0][0],
                (unsigned long long)(size_t)(const void*)kbase, DK_, 32, DK_);
    tdm_load_2d((unsigned)(size_t)(void*)&vbuf[0][0],
                (unsigned long long)(size_t)(const void*)vbase, 32, DK_, S_);
  }

  int p = 0;
  for (int kc = 0; kc < S_; kc += 32) {
    const bool more = (kc + 32) < S_;
    if (w == 0) {
      if (more) {  // prefetch next tiles into the other buffer
        tdm_load_2d((unsigned)(size_t)(void*)&kbuf[p ^ 1][0],
                    (unsigned long long)(size_t)(const void*)(kbase + (size_t)(kc + 32) * DK_),
                    DK_, 32, DK_);
        tdm_load_2d((unsigned)(size_t)(void*)&vbuf[p ^ 1][0],
                    (unsigned long long)(size_t)(const void*)(vbase + kc + 32),
                    32, DK_, S_);
        __builtin_amdgcn_s_wait_tensorcnt(2);  // current pair complete (in-order)
      } else {
        __builtin_amdgcn_s_wait_tensorcnt(0);
      }
    }
    __syncthreads();  // publish staged tiles to all waves

    // ---- S = Q @ K^T: preload all 4 K frags, then 4 back-to-back WMMAs ----
    v16bf kb[2][2];
#pragma unroll
    for (int t = 0; t < 2; ++t) {
      const unsigned short* krow = &kbuf[p][(16 * t + nn) * DK_];
      kb[t][0] = load_frag(krow, 0, half);
      kb[t][1] = load_frag(krow, 32, half);
    }
    // V frags for this step can prefetch alongside (independent of S path)
    v16bf vb[4];
#pragma unroll
    for (int c = 0; c < 4; ++c)
      vb[c] = load_frag(&vbuf[p][(16 * c + nn) * 32], 0, half);

    v8f s[2];
#pragma unroll
    for (int t = 0; t < 2; ++t) {
      v8f z = wmma_bf16(qa0, kb[t][0], zero);
      s[t] = wmma_bf16(qa1, kb[t][1], z);
    }

    // ---- mask + scale ----
    float sc[2][8];
#pragma unroll
    for (int t = 0; t < 2; ++t)
#pragma unroll
      for (int r = 0; r < 8; ++r) {
        const int mv = mrowp[r][kc + 16 * t + nn];
        sc[t][r] = mv ? s[t][r] * 0.125f : -1.0e9f;
      }

    // ---- online softmax: row stats via shfl_xor within 16-lane groups ----
    float alpha[8];
#pragma unroll
    for (int r = 0; r < 8; ++r) {
      float tm = fmaxf(sc[0][r], sc[1][r]);
      tm = fmaxf(tm, __shfl_xor(tm, 1));
      tm = fmaxf(tm, __shfl_xor(tm, 2));
      tm = fmaxf(tm, __shfl_xor(tm, 4));
      tm = fmaxf(tm, __shfl_xor(tm, 8));
      const float mnew = fmaxf(mrow[r], tm);
      alpha[r] = __expf(mrow[r] - mnew);
      mrow[r] = mnew;

      const float p0 = __expf(sc[0][r] - mnew);
      const float p1 = __expf(sc[1][r] - mnew);
      float ps = p0 + p1;
      ps += __shfl_xor(ps, 1);
      ps += __shfl_xor(ps, 2);
      ps += __shfl_xor(ps, 4);
      ps += __shfl_xor(ps, 8);
      lrow[r] = lrow[r] * alpha[r] + ps;

      // D-layout -> LDS (row = q-row-in-tile, col = local k)
      plds[w][(r + 8 * half) * 32 + nn]      = f2bf(p0);
      plds[w][(r + 8 * half) * 32 + 16 + nn] = f2bf(p1);
    }
#pragma unroll
    for (int c = 0; c < 4; ++c)
#pragma unroll
      for (int r = 0; r < 8; ++r) o[c][r] *= alpha[r];

    // within-wave DS ordering before re-reading P in A-layout
    asm volatile("s_wait_dscnt 0" ::: "memory");
    v16bf pa = load_frag(&plds[w][(size_t)nn * 32], 0, half);

    // ---- O += P @ V (4 back-to-back WMMAs, V frags already resident) ----
#pragma unroll
    for (int c = 0; c < 4; ++c) o[c] = wmma_bf16(pa, vb[c], o[c]);

    __syncthreads();  // all waves done with buffer p before TDM reuses it
    p ^= 1;
  }

  // ---- epilogue: normalize, merge heads, bf16 ctx ----
#pragma unroll
  for (int c = 0; c < 4; ++c) {
#pragma unroll
    for (int r = 0; r < 8; ++r) {
      const float val = o[c][r] / lrow[r];
      const int srow = qr0 + r + 8 * half;
      ctx[((size_t)(b * S_ + srow)) * D_ + h * DK_ + 16 * c + nn] = f2bf(val);
    }
  }
}

// ---------------------------------------------------------------------------
extern "C" void kernel_launch(void* const* d_in, const int* in_sizes, int n_in,
                              void* d_out, int out_size, void* d_ws, size_t ws_size,
                              hipStream_t stream) {
  const float* query = (const float*)d_in[0];
  const float* key   = (const float*)d_in[1];
  const float* value = (const float*)d_in[2];
  const int*   mask  = (const int*)d_in[3];
  const float* lAq = (const float*)d_in[4];
  const float* lBq = (const float*)d_in[5];
  const float* lAv = (const float*)d_in[6];
  const float* lBv = (const float*)d_in[7];
  const float* Wq = (const float*)d_in[8];
  const float* bq = (const float*)d_in[9];
  const float* Wk = (const float*)d_in[10];
  const float* bk = (const float*)d_in[11];
  const float* Wv = (const float*)d_in[12];
  const float* bv = (const float*)d_in[13];
  const float* Wm = (const float*)d_in[14];
  const float* bm = (const float*)d_in[15];

  unsigned short* ws = (unsigned short*)d_ws;
  const size_t MD = (size_t)B_ * S_ * D_;  // 3,145,728 elems
  const size_t WD = (size_t)D_ * D_;       //   589,824 elems
  unsigned short* Qp   = ws;            // LoRA-adapted query, bf16
  unsigned short* Vp   = Qp + MD;       // LoRA-adapted value, bf16
  unsigned short* keyB = Vp + MD;
  unsigned short* WqB  = keyB + MD;
  unsigned short* WkB  = WqB + WD;
  unsigned short* WvB  = WkB + WD;
  unsigned short* WmB  = WvB + WD;
  unsigned short* qhB  = WmB + WD;      // [B,H,S,DK]
  unsigned short* khB  = qhB + MD;      // [B,H,S,DK]
  unsigned short* vTB  = khB + MD;      // [B,H,DK,S]
  unsigned short* ctx  = vTB + MD;      // [B,S,D]

  convert_f32_bf16<<<512, 256, 0, stream>>>(key, keyB, MD);
  convert_f32_bf16<<<128, 256, 0, stream>>>(Wq, WqB, WD);
  convert_f32_bf16<<<128, 256, 0, stream>>>(Wk, WkB, WD);
  convert_f32_bf16<<<128, 256, 0, stream>>>(Wv, WvB, WD);
  convert_f32_bf16<<<128, 256, 0, stream>>>(Wm, WmB, WD);

  lora_adapt<<<B_ * S_, 256, 0, stream>>>(query, lAq, lBq, Qp);
  lora_adapt<<<B_ * S_, 256, 0, stream>>>(value, lAv, lBv, Vp);

  // NOTE reference's stream swap: q <- Vp@Wq^T, k <- key@Wk^T, v <- Qp@Wv^T
  dim3 gg(64, 3);  // M/64, N/256 for M=4096, N=768
  gemm_bf16_wmma<1><<<gg, 256, 0, stream>>>(Vp,   WqB, bq, qhB, B_ * S_, D_, D_);
  gemm_bf16_wmma<1><<<gg, 256, 0, stream>>>(keyB, WkB, bk, khB, B_ * S_, D_, D_);
  gemm_bf16_wmma<2><<<gg, 256, 0, stream>>>(Qp,   WvB, bv, vTB, B_ * S_, D_, D_);

  attn_wmma<<<dim3(S_ / 128, B_ * H_), 256, 0, stream>>>(qhB, khB, vTB, mask, ctx);

  gemm_bf16_wmma<0><<<gg, 256, 0, stream>>>(ctx, WmB, bm, d_out, B_ * S_, D_, D_);
}
